// SAGEBlock_40037685133650
// MI455X (gfx1250) — matrix operand order repaired and checked by reading, not compiled, not run on hardware
//
#include <hip/hip_runtime.h>
#include <math.h>

#define D 128

typedef __attribute__((ext_vector_type(16))) __bf16 v16bf;
typedef __attribute__((ext_vector_type(8)))  float  v8f;

union Frag16 {
    unsigned int u[8];
    uint4 q[2];
    v16bf v;
};

// Single-instruction f32x2 -> packed bf16 (RNE): v_cvt_pk_bf16_f32
__device__ __forceinline__ unsigned int pack_bf16(float a, float b) {
    unsigned int r;
    asm("v_cvt_pk_bf16_f32 %0, %1, %2" : "=v"(r) : "v"(a), "v"(b));
    return r;
}

// ---------------------------------------------------------------- zero ws
__global__ void __launch_bounds__(256)
sage_zero_f4(float4* __restrict__ p, long long n4) {
    long long i = (long long)blockIdx.x * blockDim.x + threadIdx.x;
    if (i < n4) p[i] = make_float4(0.f, 0.f, 0.f, 0.f);
}

// ---------------------------------------------------------------- scatter
// 32 threads per edge, 4 features each: float4 gather + 4 atomic f32 adds.
__global__ void __launch_bounds__(256)
sage_scatter(const float* __restrict__ x,
             const long long* __restrict__ src,
             const long long* __restrict__ dst,
             float* __restrict__ msg, float* __restrict__ deg,
             long long E) {
    long long gid = (long long)blockIdx.x * blockDim.x + threadIdx.x;
    long long e = gid >> 5;
    if (e >= E) return;
    int c = ((int)gid & 31) * 4;
    int s = (int)src[e];
    int d = (int)dst[e];
    const float4 v = *(const float4*)(x + (size_t)s * D + c);
    float* p = msg + (size_t)d * D + c;
    atomicAdd(p + 0, v.x);
    atomicAdd(p + 1, v.y);
    atomicAdd(p + 2, v.z);
    atomicAdd(p + 3, v.w);
    if (c == 0) atomicAdd(deg + d, 1.0f);
}

// ---------------------------------------------------------------- fused GEMM+GELU+LN+residual
// 8 waves / block; wave w handles 16 nodes. Weights staged transposed+bf16 in LDS.
// LDS layout puts each lane's 8 B-fragment dwords contiguous -> 2x ds_load_b128.
//   sW[sel][ g*1024 + j*8 + v ] = pack(W[j][2k2], W[j][2k2+1]),
//   where k2 = g*8 + v,  g = (ks*2 + hi) in 0..7,  j = output feature.
__global__ void __launch_bounds__(256)
sage_fused(const float* __restrict__ x, const float* __restrict__ msg,
           const float* __restrict__ deg,
           const float* __restrict__ Wl, const float* __restrict__ bl,
           const float* __restrict__ Wr,
           const float* __restrict__ gamma, const float* __restrict__ beta,
           float* __restrict__ out, int nNodes, int nTiles) {
    __shared__ unsigned int sW[2][64 * D];   // 64 KB exactly

    const int tid = threadIdx.x;
    for (int idx = tid; idx < 2 * 64 * D; idx += 256) {
        int sel = idx >> 13;
        int e   = idx & 8191;
        int j   = e >> 6;     // output feature 0..127
        int k2  = e & 63;     // K pair 0..63 (consecutive tid -> consecutive k2: coalesced)
        const float* Wsrc = sel ? Wr : Wl;
        float a = Wsrc[j * D + 2 * k2];
        float b = Wsrc[j * D + 2 * k2 + 1];
        sW[sel][(k2 >> 3) * 1024 + j * 8 + (k2 & 7)] = pack_bf16(a, b);
    }
    __syncthreads();

    const int wave = tid >> 5;
    const int lane = tid & 31;
    const int tile = blockIdx.x * 8 + wave;
    if (tile >= nTiles) return;            // wave-uniform: EXEC all-ones below

    const int base = tile * 16;
    const int m    = lane & 15;            // row within tile (A), col within tile (B/C)
    const int hi   = lane >> 4;

    int node  = base + m;
    int nodeL = node < nNodes ? node : nNodes - 1;   // clamp loads, mask stores later

    float dv   = deg[nodeL];
    float invd = 1.0f / (dv > 1.0f ? dv : 1.0f);

    const float* xrow = x   + (size_t)nodeL * D;
    const float* mrow = msg + (size_t)nodeL * D;

    v8f acc[8] = {};   // 8 N-tiles of 16x16 f32 accumulators

    #pragma unroll
    for (int ks = 0; ks < 4; ++ks) {       // K in steps of 32
        Frag16 aA, aX;
        const int kb = ks * 32 + hi * 16;
        #pragma unroll
        for (int v = 0; v < 4; ++v) {
            int k0 = kb + 2 * v;           // VGPR v   : K = kb+2v, kb+2v+1
            int k1 = k0 + 8;               // VGPR v+4 : K = kb+8+2v, ...
            aA.u[v]     = pack_bf16(mrow[k0] * invd, mrow[k0 + 1] * invd);
            aX.u[v]     = pack_bf16(xrow[k0],        xrow[k0 + 1]);
            aA.u[v + 4] = pack_bf16(mrow[k1] * invd, mrow[k1 + 1] * invd);
            aX.u[v + 4] = pack_bf16(xrow[k1],        xrow[k1 + 1]);
        }
        const int g = ks * 2 + hi;         // B K-group: lanes 0-15 K=0..15, 16-31 K=16..31
        #pragma unroll
        for (int t = 0; t < 8; ++t) {
            const int col = t * 16 + m;
            const uint4* pL = (const uint4*)&sW[0][g * 1024 + col * 8];
            const uint4* pR = (const uint4*)&sW[1][g * 1024 + col * 8];
            Frag16 bL, bR;
            bL.q[0] = pL[0]; bL.q[1] = pL[1];
            bR.q[0] = pR[0]; bR.q[1] = pR[1];
            acc[t] = __builtin_amdgcn_wmma_f32_16x16x32_bf16(
                false, aA.v, false, bL.v, (short)0, acc[t], false, false);
            acc[t] = __builtin_amdgcn_wmma_f32_16x16x32_bf16(
                false, aX.v, false, bR.v, (short)0, acc[t], false, false);
        }
    }

    // bias + exact GELU (in-place in accumulators)
    #pragma unroll
    for (int t = 0; t < 8; ++t) {
        const int j = t * 16 + m;
        const float bb = bl[j];
        #pragma unroll
        for (int r = 0; r < 8; ++r) {
            float f = acc[t][r] + bb;
            acc[t][r] = 0.5f * f * (1.0f + erff(f * 0.70710678118654752f));
        }
    }

    // LayerNorm per row (row M = hi*8 + r lives across 16 lanes x 8 tiles) + residual
    #pragma unroll
    for (int r = 0; r < 8; ++r) {
        float s1 = 0.f, s2 = 0.f;
        #pragma unroll
        for (int t = 0; t < 8; ++t) { float v = acc[t][r]; s1 += v; s2 += v * v; }
        #pragma unroll
        for (int off = 8; off >= 1; off >>= 1) {   // reduce within 16-lane half
            s1 += __shfl_xor(s1, off, 32);
            s2 += __shfl_xor(s2, off, 32);
        }
        const float mu   = s1 * (1.0f / 128.0f);
        const float var  = s2 * (1.0f / 128.0f) - mu * mu;
        const float rstd = rsqrtf(var + 1e-5f);

        const int rowNode = base + hi * 8 + r;
        if (rowNode < nNodes) {
            const float* xr   = x   + (size_t)rowNode * D;
            float*       orow = out + (size_t)rowNode * D;
            #pragma unroll
            for (int t = 0; t < 8; ++t) {
                const int j = t * 16 + m;
                orow[j] = (acc[t][r] - mu) * rstd * gamma[j] + beta[j] + xr[j];
            }
        }
    }
}

// ---------------------------------------------------------------- launcher
extern "C" void kernel_launch(void* const* d_in, const int* in_sizes, int n_in,
                              void* d_out, int out_size, void* d_ws, size_t ws_size,
                              hipStream_t stream) {
    const float*     x     = (const float*)d_in[0];
    const long long* ei    = (const long long*)d_in[1];   // [2, E] int64, row-major
    const float*     W_l   = (const float*)d_in[2];
    const float*     b_l   = (const float*)d_in[3];
    const float*     W_r   = (const float*)d_in[4];
    const float*     gamma = (const float*)d_in[5];
    const float*     beta  = (const float*)d_in[6];
    float*           out   = (float*)d_out;

    const int       N = in_sizes[0] / D;          // 100000
    const long long E = (long long)in_sizes[1] / 2;

    float* msg = (float*)d_ws;                    // N*128 floats
    float* deg = msg + (size_t)N * D;             // N floats

    // 1) zero msg_sum + deg
    long long n4 = ((long long)N * D + N) / 4;
    int zb = (int)((n4 + 255) / 256);
    sage_zero_f4<<<zb, 256, 0, stream>>>((float4*)d_ws, n4);

    // 2) edge scatter (mean numerator + degree)
    long long nthr = E * 32;
    int sb = (int)((nthr + 255) / 256);
    sage_scatter<<<sb, 256, 0, stream>>>(x, ei, ei + E, msg, deg, E);

    // 3) fused WMMA GEMMs + GELU + LayerNorm + residual
    int nTiles = (N + 15) / 16;
    int gb = (nTiles + 7) / 8;
    sage_fused<<<gb, 256, 0, stream>>>(x, msg, deg, W_l, b_l, W_r,
                                       gamma, beta, out, N, nTiles);
}
